// CTSimFBOnlyGLM_26448408609573
// MI455X (gfx1250) — matrix operand order repaired and checked by reading, not compiled, not run on hardware
//
#include <hip/hip_runtime.h>
#include <stdint.h>

// ---------------------------------------------------------------------------
// CTSimFBOnlyGLM on MI455X (gfx1250, wave32)
//
//  k_spatial : sp[b,t]   = sum_p sf[p] * stim[b,p,t]          (134 MB stream, BW-bound)
//  k_temporal: gen[b,l]  = bias + sum_k sp[b,l+k] * tc[k]     (tiny)
//  k_simulate: blocked stochastic recurrence. Long-range feedback per
//              16-step block is a (16 x 256) @ (256 x 16) GEMM done with
//              8 chained v_wmma_f32_16x16x32_f16; only <=15 intra-block
//              taps stay on the serial critical path.
// ---------------------------------------------------------------------------

typedef __attribute__((ext_vector_type(16))) _Float16 v16h;
typedef __attribute__((ext_vector_type(8)))  _Float16 v8h;
typedef __attribute__((ext_vector_type(8)))  float    v8f;
typedef __attribute__((ext_vector_type(4)))  float    v4f;

#define BATCH  4
#define NPIX   2048
#define NBINS  4096
#define KT     250
#define NREP   64
#define LGEN   (NBINS - KT + 1)   // 3847
#define TSIM   (LGEN - 1)         // 3846
#define BLK    16                 // timesteps per WMMA block
#define KSTEPS 8                  // 8 * 32 = 256 >= KT (K padded with zeros)
#define HWIN   1024               // circular spike-history window (bins)
#define HST    1032               // padded row stride (halves): 2064 B -> banks 4 apart

// ---------------------------------------------------------------------------
// Kernel 1: spatial projection (memory-bound streaming reduction over p)
// grid 16 x 256 threads, each thread owns 4 consecutive t (dwordx4 loads)
// ---------------------------------------------------------------------------
__global__ void __launch_bounds__(256) k_spatial(const float* __restrict__ stim,
                                                 const float* __restrict__ sf,
                                                 float* __restrict__ sp)
{
    const int b = blockIdx.x >> 2;
    const int t = ((blockIdx.x & 3) * 256 + threadIdx.x) * 4;
    const float* base = stim + ((size_t)b * NPIX) * NBINS + t;

    v4f acc = {0.f, 0.f, 0.f, 0.f};
#pragma unroll 4
    for (int p = 0; p < NPIX; ++p) {
        const float* ptr = base + (size_t)p * NBINS;
        // speculative prefetch a few rows ahead (OOB prefetch is dropped by HW)
        __builtin_prefetch(ptr + 8 * NBINS, 0, 1);
        v4f x = *(const v4f*)ptr;
        acc += x * sf[p];                // sf[p] wave-uniform -> scalar load
    }
    *(v4f*)(sp + b * NBINS + t) = acc;
}

// ---------------------------------------------------------------------------
// Kernel 2: temporal valid cross-correlation + bias
// ---------------------------------------------------------------------------
__global__ void __launch_bounds__(256) k_temporal(const float* __restrict__ sp,
                                                  const float* __restrict__ tc,
                                                  const float* __restrict__ bias,
                                                  float* __restrict__ gen)
{
    const int b = blockIdx.x >> 4;
    const int l = (blockIdx.x & 15) * 256 + threadIdx.x;
    if (l >= LGEN) return;
    const float* s = sp + b * NBINS + l;
    float acc = bias[0];
#pragma unroll 5
    for (int k = 0; k < KT; ++k)
        acc = fmaf(s[k], tc[k], acc);    // tc[k] uniform -> scalar load
    gen[b * LGEN + l] = acc;
}

// ---------------------------------------------------------------------------
// deterministic counter-based RNG (replay-stable; JAX threefry is not
// reproducible bit-exactly on-device anyway)
// ---------------------------------------------------------------------------
__device__ __forceinline__ float rng01(uint32_t bin, uint32_t strm)
{
    uint32_t h = bin * 0x9E3779B9u ^ strm * 0x85EBCA77u ^ 0x2Au;
    h ^= h >> 16; h *= 0x7FEB352Du;
    h ^= h >> 15; h *= 0x846CA68Bu;
    h ^= h >> 16;
    return (float)(h >> 8) * (1.0f / 16777216.0f);
}

// ---------------------------------------------------------------------------
// Kernel 3: blocked stochastic simulation.
// 1 wave per workgroup, 16 (b,r) streams per wave, 16 workgroups total.
// ---------------------------------------------------------------------------
__global__ void __launch_bounds__(32, 1) k_simulate(const float* __restrict__ init_spk, // (B,KT)
                                                    const float* __restrict__ fbf,      // (KT)
                                                    const float* __restrict__ gen,      // (B,LGEN)
                                                    float* __restrict__ out)            // (B,R,NBINS)
{
    __shared__ _Float16 hist[16 * HST];   // circular spike history, 16 streams
    __shared__ float    ffilt[256];       // feedback filter, zero-padded
    __shared__ float    fbL[16 * 17];     // WMMA result spill (padded)

    const int lane = threadIdx.x;         // 0..31
    const int wg   = blockIdx.x;          // 0..15
    const int b    = wg >> 2;
    const int r0   = (wg & 3) * 16;

    // ---- init LDS -----------------------------------------------------------
    {   // zero whole history window (so F==0 columns multiply clean zeros)
        uint32_t* h32 = (uint32_t*)hist;
        for (int i = lane; i < 16 * HST / 2; i += 32) h32[i] = 0u;
    }
    for (int i = lane; i < 256; i += 32)
        ffilt[i] = (i < KT) ? fbf[i] : 0.f;

    // initial 250-bin spike window -> history + output (broadcast over repeats)
    for (int i = lane; i < 16 * KT; i += 32) {
        const int s = i / KT, j = i - s * KT;
        const float v = init_spk[b * KT + j];
        hist[s * HST + j] = (_Float16)v;
        out[(size_t)(b * NREP + r0 + s) * NBINS + j] = v;
    }
    asm volatile("s_wait_dscnt 0x0" ::: "memory");

    // ---- precompute B fragments of the shifted-filter band matrix -----------
    // F[j,i] = (i <= j < KT) ? f[j-i] : 0, K padded to 256.
    // dense f16 B(32x16) layout: lanes 0-15 col=lane hold K 0..7 / 16..23; lanes 16-31 K 8..15 / 24..31.
    const int col   = lane & 15;
    const int khalf = (lane < 16) ? 0 : 8;
    v16h bfrag[KSTEPS];
#pragma unroll
    for (int kb = 0; kb < KSTEPS; ++kb) {
        v16h f;
#pragma unroll
        for (int e = 0; e < 16; ++e) {
            const int j = kb * 32 + ((e < 8) ? khalf + e : 8 + khalf + e);
            f[e] = (_Float16)((j >= col && j < KT) ? ffilt[j - col] : 0.f);
        }
        bfrag[kb] = f;
    }

    // f16 A(16x32) layout: lanes 0-15 row=lane hold K 0..7 / 16..23,
    //                      lanes 16-31 row=lane-16 hold K 8..15 / 24..31.
    const int m    = lane & 15;
    const int aoff = (lane < 16) ? 0 : 8;

    // ---- main blocked loop --------------------------------------------------
    for (int t0 = 0; t0 < TSIM; t0 += BLK) {
        const int steps = (TSIM - t0 < BLK) ? (TSIM - t0) : BLK;

        // long-range feedback: fbL[s][i] = sum_j hist[s, t0+j] * F[j,i]
        v8f acc = {};
#pragma unroll
        for (int kb = 0; kb < KSTEPS; ++kb) {
            const int ka = (t0 + kb * 32 + aoff)      & (HWIN - 1);
            const int kh = (t0 + kb * 32 + aoff + 16) & (HWIN - 1);
            v8h lo = *(const v8h*)(hist + m * HST + ka);   // ds_load_b128
            v8h hi = *(const v8h*)(hist + m * HST + kh);   // ds_load_b128
            v16h a;
#pragma unroll
            for (int e = 0; e < 8; ++e) { a[e] = lo[e]; a[8 + e] = hi[e]; }
            acc = __builtin_amdgcn_wmma_f32_16x16x32_f16(
                      false, a, false, bfrag[kb], (short)0, acc, false, false);
        }
        // spill C: VGPR v holds row v + 8*(lane>=16), col lane&15
        const int rrow = (lane >> 4) * 8;
#pragma unroll
        for (int v = 0; v < 8; ++v)
            fbL[(rrow + v) * 17 + col] = acc[v];
        asm volatile("s_wait_dscnt 0x0" ::: "memory");

        // sequential intra-block steps (<=15 serial taps each)
        for (int i = 0; i < steps; ++i) {
            const int t   = t0 + i;
            const int bin = KT + t;
            const float g = gen[b * LGEN + t];      // wave-uniform scalar load
            if (lane < 16) {
                float fb = fbL[lane * 17 + i];
                for (int j = 0; j < i; ++j)
                    fb = fmaf((float)hist[lane * HST + ((KT + t0 + j) & (HWIN - 1))],
                              ffilt[KT - (i - j)], fb);
                const float rate = 1.f / (1.f + __expf(-(g + fb)));
                const float u    = rng01((uint32_t)bin, (uint32_t)(b * NREP + r0 + lane));
                const float spk  = (u < rate) ? 1.f : 0.f;
                hist[lane * HST + (bin & (HWIN - 1))] = (_Float16)spk;
                out[(size_t)(b * NREP + r0 + lane) * NBINS + bin] = spk;
            }
            asm volatile("s_wait_dscnt 0x0" ::: "memory");
        }
    }
}

// ---------------------------------------------------------------------------
extern "C" void kernel_launch(void* const* d_in, const int* in_sizes, int n_in,
                              void* d_out, int out_size, void* d_ws, size_t ws_size,
                              hipStream_t stream)
{
    const float* stim     = (const float*)d_in[0];  // (B, NPIX, NBINS)
    const float* init_spk = (const float*)d_in[1];  // (B, KT)
    const float* sf       = (const float*)d_in[2];  // (NPIX)
    const float* tc       = (const float*)d_in[3];  // (KT)
    const float* fbf      = (const float*)d_in[4];  // (KT)
    const float* bias     = (const float*)d_in[5];  // (1)
    // d_in[6] = n_repeats (==NREP, compile-time constant here)
    (void)in_sizes; (void)n_in; (void)out_size; (void)ws_size;

    float* out = (float*)d_out;
    float* sp  = (float*)d_ws;            // B*NBINS floats
    float* gen = sp + BATCH * NBINS;      // B*LGEN  floats

    k_spatial <<<dim3(16), dim3(256), 0, stream>>>(stim, sf, sp);
    k_temporal<<<dim3(64), dim3(256), 0, stream>>>(sp, tc, bias, gen);
    k_simulate<<<dim3(16), dim3(32),  0, stream>>>(init_spk, fbf, gen, out);
}
